// Attention_26044681683510
// MI455X (gfx1250) — matrix operand order repaired and checked by reading, not compiled, not run on hardware
//
#include <hip/hip_runtime.h>
#include <hip/hip_bf16.h>
#include <math.h>

#define DIMC 1024
#define HEADS 16
#define HD 64
#define SEQ 2048
#define BATCH 2
#define MTOT (BATCH * SEQ) // 4096
#define LDSPAD 40          // 32-element rows padded to 40 shorts (80B, conflict-free)
#define KPAD 72            // 64-element rows padded to 72 shorts (144B)

typedef __attribute__((ext_vector_type(16))) __bf16 v16bf;
typedef __attribute__((ext_vector_type(8)))  float  v8f;
typedef __attribute__((ext_vector_type(8)))  unsigned short u16x8;
typedef __attribute__((ext_vector_type(16))) unsigned short u16x16;
typedef int gv4i __attribute__((vector_size(16)));   // matches builtin param type

// ---------------------------------------------------------------- helpers

__device__ __forceinline__ unsigned short f2bf(float f) {
    unsigned int u = __builtin_bit_cast(unsigned int, f);
    u += 0x7FFFu + ((u >> 16) & 1u);     // round-to-nearest-even
    return (unsigned short)(u >> 16);
}

__device__ __forceinline__ v16bf make_frag(u16x8 lo, u16x8 hi) {
    u16x16 v;
#pragma unroll
    for (int i = 0; i < 8; ++i) { v[i] = lo[i]; v[8 + i] = hi[i]; }
    return __builtin_bit_cast(v16bf, v);
}

// A fragment: 16x32 bf16 (M x K). Lanes 0-15 hold K={0..7,16..23},
// lanes 16-31 hold K={8..15,24..31}; row = lane&15.
__device__ __forceinline__ v16bf load_a_frag(const unsigned short* base,
                                             int row_stride, int lane) {
    const int r  = lane & 15;
    const int hi = lane >> 4;
    const unsigned short* p = base + (size_t)r * row_stride;
    u16x8 a = *(const u16x8*)(p + (hi ? 8 : 0));
    u16x8 b = *(const u16x8*)(p + (hi ? 24 : 16));
    return make_frag(a, b);
}

// B fragment: 32x16 bf16 (K x N) with B[k][n] = src[n*col_stride + k].
__device__ __forceinline__ v16bf load_b_frag(const unsigned short* base,
                                             int col_stride, int lane) {
    const int n  = lane & 15;
    const int hi = lane >> 4;
    const unsigned short* p = base + (size_t)n * col_stride;
    u16x8 a = *(const u16x8*)(p + (hi ? 8 : 0));
    u16x8 b = *(const u16x8*)(p + (hi ? 24 : 16));
    return make_frag(a, b);
}

__device__ __forceinline__ v8f wmma_bf16(v16bf a, v16bf b, v8f c) {
    return __builtin_amdgcn_wmma_f32_16x16x32_bf16(false, a, false, b,
                                                   (short)0, c, false, false);
}

// 16B async global->LDS copy (ASYNCcnt path); synchronous fallback if the
// builtin is unavailable on this toolchain.
__device__ __forceinline__ void async_cp16(const unsigned short* g,
                                           unsigned short* l) {
#if __has_builtin(__builtin_amdgcn_global_load_async_to_lds_b128)
    __builtin_amdgcn_global_load_async_to_lds_b128(
        (__attribute__((address_space(1))) gv4i*)(uintptr_t)g,
        (__attribute__((address_space(3))) gv4i*)(unsigned int)(uintptr_t)l,
        0, 0);
#else
    *(u16x8*)l = *(const u16x8*)g;
#endif
}

__device__ __forceinline__ void wait_async_all() {
#if __has_builtin(__builtin_amdgcn_global_load_async_to_lds_b128)
    asm volatile("s_wait_asynccnt 0x0" ::: "memory");
#endif
}

// ---------------------------------------------------------------- convert

__global__ __launch_bounds__(256) void cvt_bf16_kernel(
        const float* __restrict__ src, unsigned short* __restrict__ dst, int n) {
    int i = blockIdx.x * 256 + threadIdx.x;
    if (i < n) dst[i] = f2bf(src[i]);
}

// ------------------------------------------------- shared GEMM main loop
// Block tile 128(M) x 128(N), 8 waves of 32x64; K staged 32 at a time in
// double-buffered LDS filled by async copies (copy k+1 overlaps compute k).
__device__ __forceinline__ void gemm_mainloop_128x128(
        const unsigned short* __restrict__ Ag,
        const unsigned short* __restrict__ Bg,
        int m_blk, int n_blk,
        unsigned short* lA0, unsigned short* lA1,
        unsigned short* lB0, unsigned short* lB1,
        v8f acc[2][4]) {
    const int tid  = threadIdx.x;
    const int lane = tid & 31;
    const int wave = tid >> 5;
    const int wm = wave & 3, wn = wave >> 2;
    unsigned short* lA[2] = { lA0, lA1 };
    unsigned short* lB[2] = { lB0, lB1 };

    auto stage = [&](int buf, int k) {
#pragma unroll
        for (int i = 0; i < 2; ++i) {
            int idx = tid * 2 + i;          // 512 16B chunks per tile
            int row = idx >> 2, ch = idx & 3;
            async_cp16(Ag + (size_t)(m_blk + row) * DIMC + k + ch * 8,
                       lA[buf] + row * LDSPAD + ch * 8);
            async_cp16(Bg + (size_t)(n_blk + row) * DIMC + k + ch * 8,
                       lB[buf] + row * LDSPAD + ch * 8);
        }
    };

    stage(0, 0);
    for (int ks = 0; ks < DIMC / 32; ++ks) {
        wait_async_all();
        __syncthreads();
        if (ks + 1 < DIMC / 32) stage((ks + 1) & 1, (ks + 1) * 32);

        const unsigned short* la = lA[ks & 1] + (wm * 32) * LDSPAD;
        const unsigned short* lb = lB[ks & 1] + (wn * 64) * LDSPAD;
        v16bf a0 = load_a_frag(la,               LDSPAD, lane);
        v16bf a1 = load_a_frag(la + 16 * LDSPAD, LDSPAD, lane);
#pragma unroll
        for (int in = 0; in < 4; ++in) {
            v16bf b = load_b_frag(lb + in * 16 * LDSPAD, LDSPAD, lane);
            acc[0][in] = wmma_bf16(a0, b, acc[0][in]);
            acc[1][in] = wmma_bf16(a1, b, acc[1][in]);
        }
    }
}

// ---------------------------------------------------------------- QKV + RoPE
// grid = (1024/128, 4096/128, 3)
__global__ __launch_bounds__(256) void qkv_rope_kernel(
        const unsigned short* __restrict__ Xb,
        const unsigned short* __restrict__ Wqb,
        const unsigned short* __restrict__ Wkb,
        const unsigned short* __restrict__ Wvb,
        const float* __restrict__ cosT, const float* __restrict__ sinT,
        unsigned short* __restrict__ Qb, unsigned short* __restrict__ Kb,
        unsigned short* __restrict__ Vt) {
    __shared__ __align__(16) unsigned short ldsA[2][128 * LDSPAD];
    __shared__ __align__(16) unsigned short ldsB[2][128 * LDSPAD];

    const int lane = threadIdx.x & 31;
    const int wave = threadIdx.x >> 5;
    const int wm = wave & 3, wn = wave >> 2;
    const int which = blockIdx.z;
    const int m_blk = blockIdx.y * 128;
    const int n_blk = blockIdx.x * 128;
    const unsigned short* W = (which == 0) ? Wqb : (which == 1) ? Wkb : Wvb;

    v8f acc[2][4];
#pragma unroll
    for (int i = 0; i < 2; ++i)
#pragma unroll
        for (int j = 0; j < 4; ++j) acc[i][j] = {};

    gemm_mainloop_128x128(Xb, W, m_blk, n_blk,
                          ldsA[0], ldsA[1], ldsB[0], ldsB[1], acc);

    const int hi = lane >> 4;
    const int ln = lane & 15;
    unsigned short* dstQK = (which == 0) ? Qb : Kb;
#pragma unroll
    for (int im = 0; im < 2; ++im)
#pragma unroll
        for (int in = 0; in < 4; ++in)
#pragma unroll
            for (int vi = 0; vi < 8; ++vi) {
                float v  = acc[im][in][vi];
                float pv = __shfl_xor(v, 1, 32);          // feature c^1 partner
                int m = m_blk + wm * 32 + im * 16 + vi + 8 * hi;
                int c = n_blk + wn * 64 + in * 16 + ln;
                if (which == 2) {                          // V: transposed store
                    int b  = m >> 11, nn = m & (SEQ - 1);
                    int hh = c >> 6,  d  = c & 63;
                    Vt[(((size_t)b * HEADS + hh) * HD + d) * SEQ + nn] = f2bf(v);
                } else {                                   // Q/K: RoPE
                    int nn = m & (SEQ - 1);
                    int p  = (c & 63) >> 1;
                    float cs = cosT[nn * (HD / 2) + p];
                    float sn = sinT[nn * (HD / 2) + p];
                    float o  = (c & 1) ? (pv * sn + v * cs)
                                       : (v * cs - pv * sn);
                    dstQK[(size_t)m * DIMC + c] = f2bf(o);
                }
            }
}

// ---------------------------------------------------------------- flash attn
// One block = 8 waves sharing (b,h); each wave owns one 16-query tile.
// K/V tiles (32 keys) staged in double-buffered LDS via async copies.
// grid.x = BATCH*HEADS*(SEQ/16)/8 = 512.
__global__ __launch_bounds__(256) void flash_attn_kernel(
        const unsigned short* __restrict__ Qb,
        const unsigned short* __restrict__ Kb,
        const unsigned short* __restrict__ Vt,
        unsigned short* __restrict__ Ob) {
    __shared__ __align__(16) unsigned short Klds[2][32 * KPAD];
    __shared__ __align__(16) unsigned short Vlds[2][64 * LDSPAD];
    __shared__ __align__(16) unsigned short pbuf[8][16 * 32];

    const int tid  = threadIdx.x;
    const int lane = tid & 31;
    const int wave = tid >> 5;
    const int hi = lane >> 4, ln = lane & 15;

    const int b  = blockIdx.x >> 8;          // 16 blocks per head, 16 heads
    const int h  = (blockIdx.x >> 4) & 15;
    const int m0 = ((blockIdx.x & 15) * 8 + wave) * 16;

    const unsigned short* qbase = Qb + ((size_t)(b * SEQ + m0)) * DIMC + h * HD;
    const unsigned short* kbase = Kb + ((size_t)b * SEQ) * DIMC + h * HD;
    const unsigned short* vbase = Vt + (((size_t)b * HEADS + h) * HD) * SEQ;
    unsigned short* pl = pbuf[wave];

    v16bf Aq0 = load_a_frag(qbase,      DIMC, lane);   // d 0..31
    v16bf Aq1 = load_a_frag(qbase + 32, DIMC, lane);   // d 32..63

    v8f O[4];
#pragma unroll
    for (int t = 0; t < 4; ++t) O[t] = {};
    float mrow[8], lrow[8];
#pragma unroll
    for (int i = 0; i < 8; ++i) { mrow[i] = -3.0e38f; lrow[i] = 0.0f; }

    auto stageKV = [&](int buf, int kt) {
        int key = tid >> 3, ch = tid & 7;     // K: 32 keys x 8 chunks
        async_cp16(kbase + (size_t)(kt + key) * DIMC + ch * 8,
                   &Klds[buf][key * KPAD + ch * 8]);
        int d = tid >> 2, c = tid & 3;        // V: 64 d-rows x 4 chunks
        async_cp16(vbase + (size_t)d * SEQ + kt + c * 8,
                   &Vlds[buf][d * LDSPAD + c * 8]);
    };

    stageKV(0, 0);
    for (int it = 0; it < SEQ / 32; ++it) {
        const int cur = it & 1;
        wait_async_all();
        __syncthreads();
        if (it + 1 < SEQ / 32) stageKV((it + 1) & 1, (it + 1) * 32);

        // scores: 16 queries x 32 keys, contraction over d=64
        const unsigned short* kl = Klds[cur];
        v8f s0 = {}, s1 = {};
        s0 = wmma_bf16(Aq0, load_b_frag(kl,                KPAD, lane), s0);
        s0 = wmma_bf16(Aq1, load_b_frag(kl + 32,           KPAD, lane), s0);
        s1 = wmma_bf16(Aq0, load_b_frag(kl + 16 * KPAD,      KPAD, lane), s1);
        s1 = wmma_bf16(Aq1, load_b_frag(kl + 16 * KPAD + 32, KPAD, lane), s1);

        // online softmax per row (row = vi + 8*hi, replicated across 16 lanes)
#pragma unroll
        for (int vi = 0; vi < 8; ++vi) {
            float a  = s0[vi] * 0.125f;        // 1/sqrt(64)
            float bb = s1[vi] * 0.125f;
            float mx = fmaxf(a, bb);
#pragma unroll
            for (int msk = 1; msk < 16; msk <<= 1)
                mx = fmaxf(mx, __shfl_xor(mx, msk, 32));
            float mnew  = fmaxf(mrow[vi], mx);
            float alpha = __expf(mrow[vi] - mnew);
            float p0 = __expf(a - mnew);
            float p1 = __expf(bb - mnew);
            float sum = p0 + p1;
#pragma unroll
            for (int msk = 1; msk < 16; msk <<= 1)
                sum += __shfl_xor(sum, msk, 32);
            lrow[vi] = lrow[vi] * alpha + sum;
            mrow[vi] = mnew;
#pragma unroll
            for (int t = 0; t < 4; ++t) O[t][vi] *= alpha;
            const int r = vi + 8 * hi;
            pl[r * 32 + ln]      = f2bf(p0);
            pl[r * 32 + 16 + ln] = f2bf(p1);
        }
        asm volatile("s_wait_dscnt 0x0" ::: "memory");

        // reload P as an A fragment (16 queries x 32 keys)
        const unsigned short* pr = pl + ln * 32;
        u16x8 plo = *(const u16x8*)(pr + (hi ? 8 : 0));
        u16x8 phi = *(const u16x8*)(pr + (hi ? 24 : 16));
        v16bf Pa = make_frag(plo, phi);

        // O += P x V  (Vlds[d][key], contiguous keys per row)
        const unsigned short* vl = Vlds[cur];
#pragma unroll
        for (int t = 0; t < 4; ++t) {
            v16bf Vb = load_b_frag(vl + t * 16 * LDSPAD, LDSPAD, lane);
            O[t] = wmma_bf16(Pa, Vb, O[t]);
        }
    }

#pragma unroll
    for (int vi = 0; vi < 8; ++vi) {
        float inv = 1.0f / lrow[vi];
        int m = m0 + vi + 8 * hi;
#pragma unroll
        for (int t = 0; t < 4; ++t) {
            int d = t * 16 + ln;
            Ob[((size_t)(b * SEQ + m)) * DIMC + h * HD + d] = f2bf(O[t][vi] * inv);
        }
    }
}

// ---------------------------------------------------------------- projection
__global__ __launch_bounds__(256) void proj_kernel(
        const unsigned short* __restrict__ Ab,
        const unsigned short* __restrict__ Wpb,
        const float* __restrict__ bias, float* __restrict__ out) {
    __shared__ __align__(16) unsigned short ldsA[2][128 * LDSPAD];
    __shared__ __align__(16) unsigned short ldsB[2][128 * LDSPAD];

    const int lane = threadIdx.x & 31;
    const int wave = threadIdx.x >> 5;
    const int wm = wave & 3, wn = wave >> 2;
    const int m_blk = blockIdx.y * 128;
    const int n_blk = blockIdx.x * 128;

    v8f acc[2][4];
#pragma unroll
    for (int i = 0; i < 2; ++i)
#pragma unroll
        for (int j = 0; j < 4; ++j) acc[i][j] = {};

    gemm_mainloop_128x128(Ab, Wpb, m_blk, n_blk,
                          ldsA[0], ldsA[1], ldsB[0], ldsB[1], acc);

    const int hi = lane >> 4;
    const int ln = lane & 15;
#pragma unroll
    for (int im = 0; im < 2; ++im)
#pragma unroll
        for (int in = 0; in < 4; ++in)
#pragma unroll
            for (int vi = 0; vi < 8; ++vi) {
                int m = m_blk + wm * 32 + im * 16 + vi + 8 * hi;
                int c = n_blk + wn * 64 + in * 16 + ln;
                out[(size_t)m * DIMC + c] = acc[im][in][vi] + bias[c];
            }
}

// ---------------------------------------------------------------- launch

extern "C" void kernel_launch(void* const* d_in, const int* in_sizes, int n_in,
                              void* d_out, int out_size, void* d_ws, size_t ws_size,
                              hipStream_t stream) {
    const float* x    = (const float*)d_in[0];
    const float* wq   = (const float*)d_in[1];
    const float* wk   = (const float*)d_in[2];
    const float* wv   = (const float*)d_in[3];
    const float* wp   = (const float*)d_in[4];
    const float* bp   = (const float*)d_in[5];
    const float* cosT = (const float*)d_in[6];
    const float* sinT = (const float*)d_in[7];
    (void)in_sizes; (void)n_in; (void)out_size; (void)ws_size;

    unsigned short* ws = (unsigned short*)d_ws;
    const size_t MC = (size_t)MTOT * DIMC; // 4M elements
    const size_t CC = (size_t)DIMC * DIMC; // 1M elements
    unsigned short* Xb  = ws;
    unsigned short* Wqb = Xb + MC;
    unsigned short* Wkb = Wqb + CC;
    unsigned short* Wvb = Wkb + CC;
    unsigned short* Wpb = Wvb + CC;
    unsigned short* Qb  = Wpb + CC;
    unsigned short* Kb  = Qb + MC;
    unsigned short* Vt  = Kb + MC;
    unsigned short* Ob  = Vt + MC;

    cvt_bf16_kernel<<<(int)(MC / 256), 256, 0, stream>>>(x,  Xb,  (int)MC);
    cvt_bf16_kernel<<<(int)(CC / 256), 256, 0, stream>>>(wq, Wqb, (int)CC);
    cvt_bf16_kernel<<<(int)(CC / 256), 256, 0, stream>>>(wk, Wkb, (int)CC);
    cvt_bf16_kernel<<<(int)(CC / 256), 256, 0, stream>>>(wv, Wvb, (int)CC);
    cvt_bf16_kernel<<<(int)(CC / 256), 256, 0, stream>>>(wp, Wpb, (int)CC);

    dim3 gqkv(DIMC / 128, MTOT / 128, 3);
    qkv_rope_kernel<<<gqkv, 256, 0, stream>>>(Xb, Wqb, Wkb, Wvb,
                                              cosT, sinT, Qb, Kb, Vt);

    flash_attn_kernel<<<BATCH * HEADS * (SEQ / 16) / 8, 256, 0, stream>>>(
        Qb, Kb, Vt, Ob);

    dim3 gproj(DIMC / 128, MTOT / 128);
    proj_kernel<<<gproj, 256, 0, stream>>>(Ob, Wpb, bp, (float*)d_out);
}
